// TwoWayWordByWordAttentionLSTM_84868553769100
// MI455X (gfx1250) — compile-verified, matrix operand order checked by reading, not run on hardware
//
#include <hip/hip_runtime.h>

// ---------------------------------------------------------------------------
// TwoWayWordByWordAttentionLSTM for MI455X (gfx1250, wave32, WMMA)
//   B=256 L=64 E=300 H=100 4H=400 HID=200 V=30000 C=3
// Activations + weights staged as padded f16; every WMMA fragment is loaded
// as two contiguous 16-byte chunks (global_load_b128 / ds_load_b128).
// ---------------------------------------------------------------------------

#define BB   256
#define LL   64
#define EE   300
#define EP   320
#define HH   100
#define HP   128
#define G4   400
#define HID_ 200
#define KP2  224
#define NP_Y 112   // Npad for N=100
#define NP_1 208   // Npad for N=200
#define CC   3

typedef __attribute__((ext_vector_type(16))) _Float16 v16h;
typedef __attribute__((ext_vector_type(8)))  _Float16 v8h;
typedef __attribute__((ext_vector_type(8)))  float    v8f;

// CDNA5 ISA 7.12.2 16-bit fragment layout: for lane L, half-slots 0..7 hold
// K = laneHi*8 .. +7 and slots 8..15 hold K = 16+laneHi*8 .. +7 (contiguous).
// => one fragment = two 16-byte vector loads + a shuffle.
__device__ __forceinline__ v16h frag_ld(const _Float16* p, int laneHi) {
    const v8h lo = *(const v8h*)(p + laneHi * 8);
    const v8h hi = *(const v8h*)(p + 16 + laneHi * 8);
    return __builtin_shufflevector(lo, hi, 0, 1, 2, 3, 4, 5, 6, 7,
                                           8, 9, 10, 11, 12, 13, 14, 15);
}

__device__ __forceinline__ float sigm(float x) { return 1.0f / (1.0f + expf(-x)); }

// ---------------------------------------------------------------------------
// Pack an f32 weight matrix into padded f16 B-fragment layout Wp[n][k].
//   trans=1: Wp[n][k] = W[n*ldw + k]   (used when computing A @ W^T)
//   trans=0: Wp[n][k] = W[k*ldw + n]   (used when computing A @ W)
// ---------------------------------------------------------------------------
__global__ void pack_w(const float* __restrict__ W, int ldw, int trans,
                       int N, int K, _Float16* __restrict__ out, int Kp, int Npad)
{
    int i = blockIdx.x * blockDim.x + threadIdx.x;
    if (i >= Npad * Kp) return;
    int n = i / Kp, k = i % Kp;
    float v = 0.0f;
    if (n < N && k < K)
        v = trans ? W[(size_t)n * ldw + k] : W[(size_t)k * ldw + n];
    out[i] = (_Float16)v;
}

// ---------------------------------------------------------------------------
// Embedding gather -> padded f16 activation rows, row = t*B + b
// ---------------------------------------------------------------------------
__global__ void embed_gather(const int* __restrict__ rep,      // [B][L]
                             const float* __restrict__ emb,    // [V][E]
                             _Float16* __restrict__ Xh)        // [(t*B+b)][EP]
{
    int row = blockIdx.x;                 // t*B + b
    int t = row / BB, b = row % BB;
    int w = rep[b * LL + t];
    const float* er = emb + (size_t)w * EE;
    for (int e = threadIdx.x; e < EP; e += blockDim.x)
        Xh[(size_t)row * EP + e] = (_Float16)((e < EE) ? er[e] : 0.0f);
}

// ---------------------------------------------------------------------------
// f16 WMMA GEMM on pre-packed operands:
//   C[M,N] = A[M,Kp] x Wp[Npad,Kp]^T + bias0 + bias1, optional tanh
// One wave per 16x16 C tile; inner loop = 4x b128 loads + 1x v_wmma.
// ---------------------------------------------------------------------------
__global__ void gemm_f16_wmma(const _Float16* __restrict__ A, int lda,
                              const _Float16* __restrict__ Wp,
                              const float* __restrict__ bias0,
                              const float* __restrict__ bias1,
                              float* __restrict__ Cout, int ldc,
                              int M, int N, int Kp, int act)
{
    const int lane   = threadIdx.x & 31;
    const int wave   = threadIdx.x >> 5;
    const int tilesN = (N + 15) >> 4;
    const int tilesM = M >> 4;
    int tile = blockIdx.x * (blockDim.x >> 5) + wave;
    if (tile >= tilesM * tilesN) return;
    const int tm = tile / tilesN, tn = tile % tilesN;
    const int laneLo = lane & 15;
    const int laneHi = (lane >= 16) ? 1 : 0;
    const _Float16* arow = A  + (size_t)(tm * 16 + laneLo) * lda;
    const _Float16* brow = Wp + (size_t)(tn * 16 + laneLo) * Kp;

    v8f acc = {};
    for (int k0 = 0; k0 < Kp; k0 += 32) {
        v16h a = frag_ld(arow + k0, laneHi);
        v16h b = frag_ld(brow + k0, laneHi);
        acc = __builtin_amdgcn_wmma_f32_16x16x32_f16(false, a, false, b,
                                                     (short)0, acc, false, false);
    }
    const int bcol = tn * 16 + laneLo;
    if (bcol < N) {
        float bb = (bias0 ? bias0[bcol] : 0.0f) + (bias1 ? bias1[bcol] : 0.0f);
#pragma unroll
        for (int r = 0; r < 8; ++r) {
            int m = tm * 16 + r + (laneHi ? 8 : 0);
            float v = acc[r] + bb;
            if (act) v = tanhf(v);
            Cout[(size_t)m * ldc + bcol] = v;
        }
    }
}

// ---------------------------------------------------------------------------
// LSTM scan. One workgroup owns 16 batch rows for all 64 timesteps.
// Per step: gates = G[t] (precomputed x@Wih^T + biases) + h@Whh^T (WMMA on
// LDS-resident f16 h), then the sigmoid/tanh cell update.
// ---------------------------------------------------------------------------
__global__ void lstm_scan(const float* __restrict__ G,      // [(t*B+b)*400]
                          const _Float16* __restrict__ Whp, // packed [400][128]
                          const float* __restrict__ h0src,  // [(b*L+l)*100] or null
                          const float* __restrict__ c0src,
                          const int*   __restrict__ lenInit,// null -> zero init
                          float* __restrict__ hs,           // [(b*L+t)*100]
                          float* __restrict__ cs,           // nullable
                          _Float16* __restrict__ h16)       // nullable [(b*L+t)*128]
{
    __shared__ __attribute__((aligned(16))) _Float16 hA[16][HP];  // 4 KB
    __shared__ float gbuf[16][G4];                                // 25.6 KB
    __shared__ float cbuf[16][HH];                                // 6.4 KB
    const int tid    = threadIdx.x;
    const int lane   = tid & 31;
    const int wave   = tid >> 5;
    const int laneLo = lane & 15;
    const int laneHi = (lane >= 16) ? 1 : 0;
    const int bbase  = blockIdx.x * 16;

    for (int i = tid; i < 16 * HP; i += blockDim.x)
        hA[i >> 7][i & (HP - 1)] = (_Float16)0.0f;
    __syncthreads();
    for (int i = tid; i < 16 * HH; i += blockDim.x) {
        int bl = i / HH, hc = i % HH;
        float hv = 0.0f, cv = 0.0f;
        if (lenInit) {
            int b = bbase + bl;
            int l = lenInit[b] - 1;
            size_t ro = ((size_t)b * LL + l) * HH + hc;
            hv = h0src[ro]; cv = c0src[ro];
        }
        hA[bl][hc]   = (_Float16)hv;
        cbuf[bl][hc] = cv;
    }

    for (int t = 0; t < LL; ++t) {
        __syncthreads();                               // hA stable
        for (int tn = wave; tn < 25; tn += 8) {        // 25 N-tiles of 16
            const int ncol = tn * 16 + laneLo;
            const _Float16* brow = Whp + (size_t)ncol * HP;
            v8f acc = {};
#pragma unroll
            for (int ks = 0; ks < 4; ++ks) {           // K = 128 padded
                v16h a = frag_ld(&hA[laneLo][ks * 32], laneHi);
                v16h b = frag_ld(brow + ks * 32, laneHi);
                acc = __builtin_amdgcn_wmma_f32_16x16x32_f16(false, a, false, b,
                                                             (short)0, acc, false, false);
            }
#pragma unroll
            for (int r = 0; r < 8; ++r)
                gbuf[r + (laneHi ? 8 : 0)][ncol] = acc[r];
        }
        __syncthreads();                               // gbuf ready
        for (int e = tid; e < 16 * HH; e += blockDim.x) {
            int bl = e / HH, hc = e % HH;
            int bg = bbase + bl;
            size_t grow = ((size_t)t * BB + bg) * G4;
            if (t < LL - 1)                            // CDNA5 global_prefetch_b8
                __builtin_prefetch(G + ((size_t)(t + 1) * BB + bg) * G4 + hc, 0, 0);
            float gi = G[grow + hc]           + gbuf[bl][hc];
            float gf = G[grow + HH + hc]      + gbuf[bl][HH + hc];
            float gg = G[grow + 2 * HH + hc]  + gbuf[bl][2 * HH + hc];
            float go = G[grow + 3 * HH + hc]  + gbuf[bl][3 * HH + hc];
            float c  = sigm(gf) * cbuf[bl][hc] + sigm(gi) * tanhf(gg);
            float h  = sigm(go) * tanhf(c);
            cbuf[bl][hc] = c;
            hA[bl][hc]   = (_Float16)h;
            size_t ho = ((size_t)bg * LL + t) * HH + hc;
            hs[ho] = h;
            if (cs)  cs[ho] = c;
            if (h16) h16[((size_t)bg * LL + t) * HP + hc] = (_Float16)h;
        }
    }
}

// ---------------------------------------------------------------------------
// Recurrent attention scan; one block per batch row (rows independent).
//   wh = h2_t @ Wh + cv @ Wr ;  s_l = sum_h tanh(WyY + m_l*wh) * a_h
//   masked softmax over L ;  cv = attn-weighted sum of outp rows
// Epilogue: f = tanh(cv @ Wp + lo @ Wh) written into final[:, co:co+100].
// ---------------------------------------------------------------------------
__global__ void attn_scan(const float* __restrict__ WyY,    // [(b*L+l)*100]
                          const float* __restrict__ hs2,    // [(b*L+t)*100]
                          const float* __restrict__ outp,   // [(b*L+l)*100]
                          const float* __restrict__ maskA,  // [b*L+l]
                          const int*   __restrict__ lenB,   // [b]
                          const float* __restrict__ Wh,
                          const float* __restrict__ Wr,
                          const float* __restrict__ Wp,
                          const float* __restrict__ aW,
                          float* __restrict__ finalOut,     // [b*200 + co + n]
                          _Float16* __restrict__ finalH,    // [b*224 + co + n]
                          int co)
{
    __shared__ float cv[HH], wh[HH], sv[LL], ev[LL], red[1];
    const int b = blockIdx.x, tid = threadIdx.x;
    if (tid < HH) cv[tid] = 0.0f;
    __syncthreads();
    const int tlast = lenB[b] - 1;
    for (int t = 0;; ++t) {
        if (tid < HH) {
            const float* hrow = hs2 + ((size_t)b * LL + t) * HH;
            float s = 0.0f;
            for (int k = 0; k < HH; ++k)
                s += hrow[k] * Wh[k * HH + tid] + cv[k] * Wr[k * HH + tid];
            wh[tid] = s;
        }
        __syncthreads();
        if (tid < LL) {
            const float m = maskA[b * LL + tid];
            const float* yrow = WyY + ((size_t)b * LL + tid) * HH;
            float s = 0.0f;
            for (int h = 0; h < HH; ++h)
                s += tanhf(yrow[h] + m * wh[h]) * aW[h];
            sv[tid] = s;
        }
        __syncthreads();
        if (tid == 0) {
            float mx = sv[0];
            for (int l = 1; l < LL; ++l) mx = fmaxf(mx, sv[l]);
            float sum = 0.0f;
            for (int l = 0; l < LL; ++l) {
                float e = expf(sv[l] - mx) * maskA[b * LL + l];
                ev[l] = e; sum += e;
            }
            red[0] = 1.0f / sum;
        }
        __syncthreads();
        if (tid < HH) {
            const float inv = red[0];
            float s = 0.0f;
            for (int l = 0; l < LL; ++l)
                s += ev[l] * inv * outp[((size_t)b * LL + l) * HH + tid];
            cv[tid] = s;
        }
        __syncthreads();
        if (t == tlast) break;
    }
    if (tid < HH) {
        const float* lo = hs2 + ((size_t)b * LL + tlast) * HH;
        float s = 0.0f;
        for (int k = 0; k < HH; ++k)
            s += cv[k] * Wp[k * HH + tid] + lo[k] * Wh[k * HH + tid];
        float f = tanhf(s);
        finalOut[(size_t)b * (2 * HH) + co + tid] = f;
        finalH[(size_t)b * KP2 + co + tid] = (_Float16)f;
    }
}

// ---------------------------------------------------------------------------
// Final tiny head: logits = fc @ W2^T + b2 (C=3) then log_softmax per row.
// ---------------------------------------------------------------------------
__global__ void fc2_logsoftmax(const float* __restrict__ fc,  // [256][200]
                               const float* __restrict__ W2,  // [3][200]
                               const float* __restrict__ b2,  // [3]
                               float* __restrict__ out)       // [256][3]
{
    int b = blockIdx.x * blockDim.x + threadIdx.x;
    if (b >= BB) return;
    float lg[CC];
    const float* x = fc + (size_t)b * HID_;
    for (int c = 0; c < CC; ++c) {
        const float* w = W2 + (size_t)c * HID_;
        float s = b2[c];
        for (int k = 0; k < HID_; ++k) s += x[k] * w[k];
        lg[c] = s;
    }
    float mx = fmaxf(lg[0], fmaxf(lg[1], lg[2]));
    float lse = logf(expf(lg[0] - mx) + expf(lg[1] - mx) + expf(lg[2] - mx));
    for (int c = 0; c < CC; ++c) out[b * CC + c] = lg[c] - mx - lse;
}

// ---------------------------------------------------------------------------
extern "C" void kernel_launch(void* const* d_in, const int* in_sizes, int n_in,
                              void* d_out, int out_size, void* d_ws, size_t ws_size,
                              hipStream_t stream) {
    (void)in_sizes; (void)n_in; (void)out_size; (void)ws_size;
    const int*   rep1  = (const int*)  d_in[0];
    const int*   len1  = (const int*)  d_in[1];
    const float* mask1 = (const float*)d_in[2];
    const int*   rep2  = (const int*)  d_in[3];
    const int*   len2  = (const int*)  d_in[4];
    const float* mask2 = (const float*)d_in[5];
    const float* emb   = (const float*)d_in[6];
    const float* W_ih1 = (const float*)d_in[7];
    const float* W_hh1 = (const float*)d_in[8];
    const float* b_ih1 = (const float*)d_in[9];
    const float* b_hh1 = (const float*)d_in[10];
    const float* W_ih2 = (const float*)d_in[11];
    const float* W_hh2 = (const float*)d_in[12];
    const float* b_ih2 = (const float*)d_in[13];
    const float* b_hh2 = (const float*)d_in[14];
    const float* Wy    = (const float*)d_in[15];
    const float* Wh    = (const float*)d_in[16];
    const float* Wr    = (const float*)d_in[17];
    const float* Wp    = (const float*)d_in[18];
    const float* aW    = (const float*)d_in[19];
    const float* W1    = (const float*)d_in[20];
    const float* b1    = (const float*)d_in[21];
    const float* W2    = (const float*)d_in[22];
    const float* b2    = (const float*)d_in[23];
    float* out = (float*)d_out;

    char* ws = (char*)d_ws;
    size_t off = 0;
    auto alloc = [&](size_t bytes) -> char* {
        char* p = ws + off;
        off = (off + bytes + 255) & ~(size_t)255;
        return p;
    };
    const size_t ROWS = (size_t)LL * BB;               // 16384
    _Float16* X1h    = (_Float16*)alloc(ROWS * EP * 2);
    _Float16* X2h    = (_Float16*)alloc(ROWS * EP * 2);
    float*    G1     = (float*)   alloc(ROWS * G4 * 4);
    float*    G2     = (float*)   alloc(ROWS * G4 * 4);
    float*    hs1    = (float*)   alloc(ROWS * HH * 4);
    float*    cs1    = (float*)   alloc(ROWS * HH * 4);
    float*    hs2    = (float*)   alloc(ROWS * HH * 4);
    _Float16* h16    = (_Float16*)alloc(ROWS * HP * 2);
    float*    WyYb   = (float*)   alloc(ROWS * HH * 4);
    float*    finalb = (float*)   alloc((size_t)BB * 2 * HH * 4);
    _Float16* finalh = (_Float16*)alloc((size_t)BB * KP2 * 2);
    float*    fc     = (float*)   alloc((size_t)BB * HID_ * 4);
    _Float16* Wih1p  = (_Float16*)alloc((size_t)G4 * EP * 2);
    _Float16* Wih2p  = (_Float16*)alloc((size_t)G4 * EP * 2);
    _Float16* Whh1p  = (_Float16*)alloc((size_t)G4 * HP * 2);
    _Float16* Whh2p  = (_Float16*)alloc((size_t)G4 * HP * 2);
    _Float16* Wyp    = (_Float16*)alloc((size_t)NP_Y * HP * 2);
    _Float16* W1p    = (_Float16*)alloc((size_t)NP_1 * KP2 * 2);

    hipMemsetAsync(h16,    0, ROWS * HP * 2,        stream);  // zero K-pad cols
    hipMemsetAsync(finalh, 0, (size_t)BB * KP2 * 2, stream);

    // weight packing (padded f16, fragment-friendly K-contiguous layout)
    pack_w<<<(G4 * EP + 255) / 256, 256, 0, stream>>>(W_ih1, EE, 1, G4, EE, Wih1p, EP, G4);
    pack_w<<<(G4 * EP + 255) / 256, 256, 0, stream>>>(W_ih2, EE, 1, G4, EE, Wih2p, EP, G4);
    pack_w<<<(G4 * HP + 255) / 256, 256, 0, stream>>>(W_hh1, HH, 1, G4, HH, Whh1p, HP, G4);
    pack_w<<<(G4 * HP + 255) / 256, 256, 0, stream>>>(W_hh2, HH, 1, G4, HH, Whh2p, HP, G4);
    pack_w<<<(NP_Y * HP + 255) / 256, 256, 0, stream>>>(Wy, HH, 0, HH, HH, Wyp, HP, NP_Y);
    pack_w<<<(NP_1 * KP2 + 255) / 256, 256, 0, stream>>>(W1, HID_, 1, HID_, HID_, W1p, KP2, NP_1);

    embed_gather<<<(int)ROWS, 256, 0, stream>>>(rep1, emb, X1h);
    embed_gather<<<(int)ROWS, 256, 0, stream>>>(rep2, emb, X2h);

    for (int dir = 0; dir < 2; ++dir) {
        const _Float16* XA = dir ? X2h : X1h;          // sen1 of this direction
        const _Float16* XB = dir ? X1h : X2h;          // sen2
        const float* maskA = dir ? mask2 : mask1;
        const int*   lenA  = dir ? len2  : len1;
        const int*   lenB  = dir ? len1  : len2;

        // gate precompute: [16384 x 400] = XA[16384x320] @ W_ih^T + b_ih + b_hh
        gemm_f16_wmma<<<3200, 256, 0, stream>>>(XA, EP, Wih1p, b_ih1, b_hh1,
                                                G1, G4, (int)ROWS, G4, EP, 0);
        gemm_f16_wmma<<<3200, 256, 0, stream>>>(XB, EP, Wih2p, b_ih2, b_hh2,
                                                G2, G4, (int)ROWS, G4, EP, 0);
        // LSTM1 from zero state; LSTM2 seeded with (h,c) at t = lenA-1
        lstm_scan<<<BB / 16, 256, 0, stream>>>(G1, Whh1p, nullptr, nullptr,
                                               nullptr, hs1, cs1, h16);
        lstm_scan<<<BB / 16, 256, 0, stream>>>(G2, Whh2p, hs1, cs1,
                                               lenA, hs2, nullptr, nullptr);
        // WyY = outp @ Wy
        gemm_f16_wmma<<<896, 256, 0, stream>>>(h16, HP, Wyp, nullptr, nullptr,
                                               WyYb, HH, (int)ROWS, HH, HP, 0);
        attn_scan<<<BB, 128, 0, stream>>>(WyYb, hs2, hs1, maskA, lenB,
                                          Wh, Wr, Wp, aW,
                                          finalb, finalh, dir * HH);
    }
    // fc = tanh(final @ W1^T + b1)
    gemm_f16_wmma<<<26, 256, 0, stream>>>(finalh, KP2, W1p, b1, nullptr,
                                          fc, HID_, BB, HID_, KP2, 1);
    fc2_logsoftmax<<<1, 256, 0, stream>>>(fc, W2, b2, out);
}
